// CrossAttention_37495064494692
// MI455X (gfx1250) — matrix operand order repaired and checked by reading, not compiled, not run on hardware
//
#include <hip/hip_runtime.h>
#include <hip/hip_bf16.h>
#include <math.h>

typedef __attribute__((ext_vector_type(16))) __bf16 v16bf;
typedef __attribute__((ext_vector_type(8)))  __bf16 v8bf;
typedef __attribute__((ext_vector_type(4)))  __bf16 v4bf;
typedef __attribute__((ext_vector_type(8)))  float  v8f;

#define DEV __device__ __forceinline__

namespace {

constexpr int Bb = 8, Ss = 2048, Ll = 512, Ee = 1024, TEe = 768, Hh = 16, Dd = 64;
constexpr float kScale = 0.125f;  // D^-0.5 = 64^-0.5

DEV v8f wmma_bf16(v16bf a, v16bf b, v8f c) {
  // 8 args: (neg_a, A, neg_b, B, c_mod, C, reuse_a, reuse_b)
  return __builtin_amdgcn_wmma_f32_16x16x32_bf16(false, a, false, b, (short)0, c,
                                                 false, false);
}

DEV v8f zero8() {
  v8f z;
#pragma unroll
  for (int i = 0; i < 8; ++i) z[i] = 0.f;
  return z;
}

// A fragment (16x32 bf16, MxK) per CDNA5 WMMA layout:
// lane = {kg, m[3:0]}; VGPR0-3: K = kg*8 + [0..7]; VGPR4-7: K = 16 + kg*8 + [0..7]
DEV v16bf load_frag_a(const __bf16* tile, int ld) {
  const int lane = threadIdx.x & 31;
  const int m = lane & 15, kg = lane >> 4;
  const __bf16* p = tile + m * ld + kg * 8;
  v8bf lo = *(const v8bf*)(p);
  v8bf hi = *(const v8bf*)(p + 16);
  return __builtin_shufflevector(lo, hi, 0, 1, 2, 3, 4, 5, 6, 7,
                                 8, 9, 10, 11, 12, 13, 14, 15);
}

// B fragment (32x16 bf16, KxN) sourced from an N-major (transposed) tile:
// tileT[n][k]; lane = {kg, n[3:0]}; VGPR0-3: K = kg*8 + [0..7]; VGPR4-7: K = 16 + kg*8 + [0..7]
DEV v16bf load_frag_b(const __bf16* tileT, int ld) {
  const int lane = threadIdx.x & 31;
  const int n = lane & 15, kg = lane >> 4;
  const __bf16* p = tileT + n * ld + kg * 8;
  v8bf lo = *(const v8bf*)(p);
  v8bf hi = *(const v8bf*)(p + 16);
  return __builtin_shufflevector(lo, hi, 0, 1, 2, 3, 4, 5, 6, 7,
                                 8, 9, 10, 11, 12, 13, 14, 15);
}

__global__ __launch_bounds__(256) void cvt_f32_bf16_kernel(
    const float* __restrict__ x, __bf16* __restrict__ y, int n4) {
  int i = blockIdx.x * blockDim.x + threadIdx.x;
  if (i >= n4) return;
  float4 f = ((const float4*)x)[i];
  v4bf o = {(__bf16)f.x, (__bf16)f.y, (__bf16)f.z, (__bf16)f.w};
  ((v4bf*)y)[i] = o;
}

// C[M,N] = A[M,K] @ W[K,N] + bias, bf16 operands, fp32 accumulate.
// Block tile 128(M) x 128(N), K step 32, double-buffered LDS.
// 8 waves, each owning a 32x64 tile (2x4 WMMA tiles -> 8 WMMA / K-step).
template <bool OUT_F32>
__global__ __launch_bounds__(256) void gemm_bf16_kernel(
    const __bf16* __restrict__ A, const __bf16* __restrict__ W,
    const float* __restrict__ bias, void* __restrict__ C,
    int M, int N, int K, int ldc) {
  __shared__ __attribute__((aligned(16))) __bf16 a_lds[2][128 * 32];
  __shared__ __attribute__((aligned(16))) __bf16 bT_lds[2][128 * 32];

  const int t = threadIdx.x;
  const int m0 = blockIdx.x * 128;
  const int n0 = blockIdx.y * 128;
  const int wave = t >> 5;
  const int wm = (wave >> 1) * 32;  // 0,32,64,96
  const int wn = (wave & 1) * 64;   // 0,64

  v8f acc[2][4];
#pragma unroll
  for (int mi = 0; mi < 2; ++mi)
#pragma unroll
    for (int ni = 0; ni < 4; ++ni) acc[mi][ni] = zero8();

  // Stage one K-slab (A: 128x32 row-major, W: 32x128 transposed to [n][k]).
  auto stage = [&](int buf, int k0) {
    __bf16* al = &a_lds[buf][0];
    __bf16* bl = &bT_lds[buf][0];
#pragma unroll
    for (int s2 = 0; s2 < 2; ++s2) {
      int s = t + s2 * 256;
      int ar = s >> 2, ac = (s & 3) * 8;
      *(v8bf*)&al[ar * 32 + ac] =
          *(const v8bf*)&A[(size_t)(m0 + ar) * K + k0 + ac];
      int kr = s >> 4, nc = (s & 15) * 8;
      v8bf vb = *(const v8bf*)&W[(size_t)(k0 + kr) * N + n0 + nc];
#pragma unroll
      for (int i = 0; i < 8; ++i) bl[(nc + i) * 32 + kr] = vb[i];
    }
  };

  stage(0, 0);
  int buf = 0;
  for (int k0 = 0; k0 < K; k0 += 32, buf ^= 1) {
    __syncthreads();  // staged slab visible; previous slab's readers done
    if (k0 + 32 < K) stage(buf ^ 1, k0 + 32);  // overlap next slab with WMMAs

    v16bf af[2], bfr[4];
#pragma unroll
    for (int mi = 0; mi < 2; ++mi)
      af[mi] = load_frag_a(&a_lds[buf][(wm + mi * 16) * 32], 32);
#pragma unroll
    for (int ni = 0; ni < 4; ++ni)
      bfr[ni] = load_frag_b(&bT_lds[buf][(wn + ni * 16) * 32], 32);
#pragma unroll
    for (int mi = 0; mi < 2; ++mi)
#pragma unroll
      for (int ni = 0; ni < 4; ++ni)
        acc[mi][ni] = wmma_bf16(af[mi], bfr[ni], acc[mi][ni]);
  }

  const int lane = t & 31, nlo = lane & 15, hi2 = lane >> 4;
#pragma unroll
  for (int mi = 0; mi < 2; ++mi)
#pragma unroll
    for (int ni = 0; ni < 4; ++ni) {
      int col = n0 + wn + ni * 16 + nlo;
      float bv = bias[col];
#pragma unroll
      for (int r = 0; r < 8; ++r) {
        int row = m0 + wm + mi * 16 + r + 8 * hi2;
        float v = acc[mi][ni][r] + bv;
        if (OUT_F32)
          ((float*)C)[(size_t)row * ldc + col] = v;
        else
          ((__bf16*)C)[(size_t)row * ldc + col] = (__bf16)v;
      }
    }
}

// Fused flash attention: block = 128 query rows of one (b,h); L swept in
// 128-chunks with online softmax. All matmuls via WMMA bf16.
__global__ __launch_bounds__(256) void flash_attn_kernel(
    const __bf16* __restrict__ q,   // [B*S, E]  (col = h*64 + d)
    const __bf16* __restrict__ kv,  // [B*L, 2E] (k: h*64+d, v: E + h*64+d)
    const int* __restrict__ amask,  // [B, L]
    __bf16* __restrict__ out) {     // [B*S, E]
  __shared__ __attribute__((aligned(16))) __bf16 q_lds[128 * 64];
  __shared__ __attribute__((aligned(16))) __bf16 k_lds[128 * 64];   // [l][d]
  __shared__ __attribute__((aligned(16))) __bf16 vT_lds[64 * 128];  // [d][l]
  __shared__ __attribute__((aligned(16))) __bf16 p_lds[128 * 128];
  __shared__ float mneg[128];

  const int t = threadIdx.x;
  const int s0 = blockIdx.x * 128;
  const int b = blockIdx.y >> 4;  // / H
  const int h = blockIdx.y & 15;
  const int wave = t >> 5;
  const int lane = t & 31, nlo = lane & 15, hi2 = lane >> 4;

  // stage Q tile 128x64
#pragma unroll
  for (int s2 = 0; s2 < 4; ++s2) {
    int slot = t + s2 * 256;
    int r = slot >> 3, c = (slot & 7) * 8;
    *(v8bf*)&q_lds[r * 64 + c] =
        *(const v8bf*)&q[(size_t)(b * Ss + s0 + r) * Ee + h * 64 + c];
  }

  float rM[8], rS[8];
#pragma unroll
  for (int r = 0; r < 8; ++r) { rM[r] = -3.0e38f; rS[r] = 0.f; }
  v8f oacc[4];
#pragma unroll
  for (int ot = 0; ot < 4; ++ot) oacc[ot] = zero8();

  __syncthreads();
  v16bf qf[2];
#pragma unroll
  for (int ks = 0; ks < 2; ++ks)
    qf[ks] = load_frag_a(&q_lds[(wave * 16) * 64 + ks * 32], 64);

  for (int c = 0; c < Ll / 128; ++c) {
    const int l0 = c * 128;
    __syncthreads();  // previous chunk's k/v/p consumers done
    // stage K tile [l][d] and V tile transposed [d][l]
#pragma unroll
    for (int s2 = 0; s2 < 4; ++s2) {
      int slot = t + s2 * 256;
      int r = slot >> 3, cc = (slot & 7) * 8;
      *(v8bf*)&k_lds[r * 64 + cc] =
          *(const v8bf*)&kv[(size_t)(b * Ll + l0 + r) * (2 * Ee) + h * 64 + cc];
      v8bf vv = *(const v8bf*)&kv[(size_t)(b * Ll + l0 + r) * (2 * Ee) + Ee +
                                  h * 64 + cc];
#pragma unroll
      for (int i = 0; i < 8; ++i) vT_lds[(cc + i) * 128 + r] = vv[i];
    }
    if (t < 128) mneg[t] = (amask[b * Ll + l0 + t] == 0) ? -3.0e38f : 0.f;
    if (c + 1 < Ll / 128 && t < 128)
      __builtin_prefetch(
          &kv[(size_t)(b * Ll + l0 + 128 + t) * (2 * Ee) + h * 64], 0, 0);
    __syncthreads();

    // S = Q x K^T  (16 rows x 128 cols per wave)
    v8f sacc[8];
#pragma unroll
    for (int nt = 0; nt < 8; ++nt) {
      sacc[nt] = zero8();
#pragma unroll
      for (int ks = 0; ks < 2; ++ks) {
        v16bf kf = load_frag_b(&k_lds[(nt * 16) * 64 + ks * 32], 64);
        sacc[nt] = wmma_bf16(qf[ks], kf, sacc[nt]);
      }
    }

    // scale + mask + chunk row-max
    float cmax[8];
#pragma unroll
    for (int r = 0; r < 8; ++r) cmax[r] = -3.0e38f;
#pragma unroll
    for (int nt = 0; nt < 8; ++nt) {
      float mv = mneg[nt * 16 + nlo];
#pragma unroll
      for (int r = 0; r < 8; ++r) {
        float x = sacc[nt][r] * kScale + mv;
        sacc[nt][r] = x;
        cmax[r] = fmaxf(cmax[r], x);
      }
    }
#pragma unroll
    for (int off = 1; off <= 8; off <<= 1)
#pragma unroll
      for (int r = 0; r < 8; ++r)
        cmax[r] = fmaxf(cmax[r], __shfl_xor(cmax[r], off, 32));

    float alpha[8];
#pragma unroll
    for (int r = 0; r < 8; ++r) {
      float nm = fmaxf(rM[r], cmax[r]);
      alpha[r] = __expf(rM[r] - nm);
      rM[r] = nm;
    }
    float csum[8];
#pragma unroll
    for (int r = 0; r < 8; ++r) csum[r] = 0.f;
#pragma unroll
    for (int nt = 0; nt < 8; ++nt)
#pragma unroll
      for (int r = 0; r < 8; ++r) {
        float p = __expf(sacc[nt][r] - rM[r]);
        sacc[nt][r] = p;
        csum[r] += p;
      }
#pragma unroll
    for (int off = 1; off <= 8; off <<= 1)
#pragma unroll
      for (int r = 0; r < 8; ++r) csum[r] += __shfl_xor(csum[r], off, 32);
#pragma unroll
    for (int r = 0; r < 8; ++r) rS[r] = rS[r] * alpha[r] + csum[r];
#pragma unroll
    for (int ot = 0; ot < 4; ++ot)
#pragma unroll
      for (int r = 0; r < 8; ++r) oacc[ot][r] *= alpha[r];

    // spill P (bf16) into wave-private LDS rows, re-read as A fragments
#pragma unroll
    for (int nt = 0; nt < 8; ++nt)
#pragma unroll
      for (int r = 0; r < 8; ++r)
        p_lds[(wave * 16 + r + 8 * hi2) * 128 + nt * 16 + nlo] =
            (__bf16)sacc[nt][r];
    asm volatile("s_wait_dscnt 0" ::: "memory");

    // O += P x V
#pragma unroll
    for (int ks = 0; ks < 4; ++ks) {
      v16bf pf = load_frag_a(&p_lds[(wave * 16) * 128 + ks * 32], 128);
#pragma unroll
      for (int ot = 0; ot < 4; ++ot) {
        v16bf vf = load_frag_b(&vT_lds[(ot * 16) * 128 + ks * 32], 128);
        oacc[ot] = wmma_bf16(pf, vf, oacc[ot]);
      }
    }
  }

  // normalize and store merged-head layout [B*S, E]
#pragma unroll
  for (int ot = 0; ot < 4; ++ot)
#pragma unroll
    for (int r = 0; r < 8; ++r) {
      float o = oacc[ot][r] / rS[r];
      int row = b * Ss + s0 + wave * 16 + r + 8 * hi2;
      out[(size_t)row * Ee + h * 64 + ot * 16 + nlo] = (__bf16)o;
    }
}

}  // namespace

extern "C" void kernel_launch(void* const* d_in, const int* in_sizes, int n_in,
                              void* d_out, int out_size, void* d_ws,
                              size_t ws_size, hipStream_t stream) {
  (void)in_sizes; (void)n_in; (void)out_size; (void)ws_size;
  const float* query = (const float*)d_in[0];
  const float* guide = (const float*)d_in[1];
  const int* amask = (const int*)d_in[2];
  const float* Wt = (const float*)d_in[3];
  const float* bt = (const float*)d_in[4];
  const float* Wq = (const float*)d_in[5];
  const float* bq = (const float*)d_in[6];
  const float* Wkv = (const float*)d_in[7];
  const float* bkv = (const float*)d_in[8];
  const float* Wo = (const float*)d_in[9];
  const float* bo = (const float*)d_in[10];
  float* out = (float*)d_out;

  char* ws = (char*)d_ws;
  size_t off = 0;
  auto alloc = [&](size_t bytes) -> void* {
    void* p = ws + off;
    off += (bytes + 255) & ~(size_t)255;
    return p;
  };
  __bf16* qry_b = (__bf16*)alloc((size_t)Bb * Ss * Ee * 2);
  __bf16* gui_b = (__bf16*)alloc((size_t)Bb * Ll * TEe * 2);
  __bf16* Wt_b  = (__bf16*)alloc((size_t)TEe * Ee * 2);
  __bf16* Wq_b  = (__bf16*)alloc((size_t)Ee * Ee * 2);
  __bf16* Wkv_b = (__bf16*)alloc((size_t)Ee * 2 * Ee * 2);
  __bf16* Wo_b  = (__bf16*)alloc((size_t)Ee * Ee * 2);
  __bf16* g_b   = (__bf16*)alloc((size_t)Bb * Ll * Ee * 2);
  __bf16* kv_b  = (__bf16*)alloc((size_t)Bb * Ll * 2 * Ee * 2);
  __bf16* q_b   = (__bf16*)alloc((size_t)Bb * Ss * Ee * 2);
  __bf16* ao_b  = (__bf16*)alloc((size_t)Bb * Ss * Ee * 2);

  auto cvt = [&](const float* src, __bf16* dst, size_t n) {
    int n4 = (int)(n / 4);
    cvt_f32_bf16_kernel<<<(n4 + 255) / 256, 256, 0, stream>>>(src, dst, n4);
  };
  cvt(query, qry_b, (size_t)Bb * Ss * Ee);
  cvt(guide, gui_b, (size_t)Bb * Ll * TEe);
  cvt(Wt, Wt_b, (size_t)TEe * Ee);
  cvt(Wq, Wq_b, (size_t)Ee * Ee);
  cvt(Wkv, Wkv_b, (size_t)Ee * 2 * Ee);
  cvt(Wo, Wo_b, (size_t)Ee * Ee);

  // g = guide @ Wt + bt           [4096 x 1024]
  gemm_bf16_kernel<false><<<dim3((Bb * Ll) / 128, Ee / 128), 256, 0, stream>>>(
      gui_b, Wt_b, bt, g_b, Bb * Ll, Ee, TEe, Ee);
  // q = query @ Wq + bq           [16384 x 1024]
  gemm_bf16_kernel<false><<<dim3((Bb * Ss) / 128, Ee / 128), 256, 0, stream>>>(
      qry_b, Wq_b, bq, q_b, Bb * Ss, Ee, Ee, Ee);
  // kv = g @ Wkv + bkv            [4096 x 2048]
  gemm_bf16_kernel<false><<<dim3((Bb * Ll) / 128, (2 * Ee) / 128), 256, 0, stream>>>(
      g_b, Wkv_b, bkv, kv_b, Bb * Ll, 2 * Ee, Ee, 2 * Ee);
  // fused attention               -> ao [16384 x 1024]
  flash_attn_kernel<<<dim3(Ss / 128, Bb * Hh), 256, 0, stream>>>(q_b, kv_b,
                                                                 amask, ao_b);
  // out = ao @ Wo + bo (fp32)     [16384 x 1024]
  gemm_bf16_kernel<true><<<dim3((Bb * Ss) / 128, Ee / 128), 256, 0, stream>>>(
      ao_b, Wo_b, bo, out, Bb * Ss, Ee, Ee, Ee);
}
